// CausalSelfAttention_20985210208905
// MI455X (gfx1250) — compile-verified, matrix operand order checked by reading, not run on hardware
//
#include <hip/hip_runtime.h>

// B=2, T=2048, C=1024, H=16, HD=64
#define BB 2
#define TT 2048
#define CC 1024
#define HH 16
#define HD 64

typedef __attribute__((ext_vector_type(16))) __bf16 v16bf;
typedef __attribute__((ext_vector_type(8)))  __bf16 bf16x8;
typedef __attribute__((ext_vector_type(8)))  float  v8f;

static __device__ inline __bf16 cvt_bf16(float f) {
    unsigned u = __builtin_bit_cast(unsigned, f);
    unsigned r = u + 0x7fffu + ((u >> 16) & 1u);   // round-to-nearest-even
    unsigned short h = (unsigned short)(r >> 16);
    return __builtin_bit_cast(__bf16, h);
}

// bf16 WMMA fragment (A or B role): per lane, 8 contiguous elems at p and 8 at
// p+16, per the CDNA5 16-bit 16x32 VGPR layout.
static __device__ inline v16bf frag16(const __bf16* p) {
    bf16x8 lo = *reinterpret_cast<const bf16x8*>(p);
    bf16x8 hi = *reinterpret_cast<const bf16x8*>(p + 16);
    return __builtin_shufflevector(lo, hi, 0,1,2,3,4,5,6,7,8,9,10,11,12,13,14,15);
}

static __device__ inline v8f wmma_bf16(v16bf a, v16bf b, v8f c) {
    return __builtin_amdgcn_wmma_f32_16x16x32_bf16(false, a, false, b, (short)0, c, false, false);
}

// ---------------------------------------------------------------- convert
__global__ void csa_cvt_f32_bf16(const float* __restrict__ src,
                                 __bf16* __restrict__ dst, int n) {
    int i = blockIdx.x * blockDim.x + threadIdx.x;
    if (i < n) dst[i] = cvt_bf16(src[i]);
}

// ------------------------------------------------------- GEMM wave helpers
// One wave computes a 32x64 output tile: fr = {A0, A1, B0..B3}.
static __device__ inline void load_frags6(v16bf fr[6],
                                          const __bf16* __restrict__ arow0,
                                          const __bf16* __restrict__ arow1,
                                          const __bf16* __restrict__ wrow,
                                          int k) {
    fr[0] = frag16(arow0 + k);
    fr[1] = frag16(arow1 + k);
#pragma unroll
    for (int nt = 0; nt < 4; ++nt)
        fr[2 + nt] = frag16(wrow + (size_t)nt * 16 * CC + k);
}

static __device__ inline void wmma_tile8(const v16bf fr[6], v8f acc[8]) {
#pragma unroll
    for (int mi = 0; mi < 2; ++mi)
#pragma unroll
        for (int nt = 0; nt < 4; ++nt)
            acc[mi * 4 + nt] = wmma_bf16(fr[mi], fr[2 + nt], acc[mi * 4 + nt]);
}

// ---------------------------------------------------------------- QKV GEMM
// out[m,n] = sum_k x[m,k] * W[n,k] + bias[n]   (x@W^T)
// p==0 -> Q [B,H,T,HD];  p==1 -> K [B,H,T,HD];  p==2 -> V transposed [B,H,HD,T]
__global__ __launch_bounds__(256) void csa_qkv_proj(
    const __bf16* __restrict__ xb,
    const __bf16* __restrict__ wq, const __bf16* __restrict__ wk,
    const __bf16* __restrict__ wv,
    const float* __restrict__ bq, const float* __restrict__ bk,
    const float* __restrict__ bv,
    __bf16* __restrict__ qo, __bf16* __restrict__ ko, __bf16* __restrict__ vto)
{
    const int lane = threadIdx.x & 31;
    const int wid  = threadIdx.x >> 5;
    const int ln   = lane & 15;
    const int kb8  = (lane >> 4) * 8;
    const int half = (lane >> 4) << 3;

    const int tile = blockIdx.x * 8 + wid;      // 2048 tiles: 128 m x 16 n
    const int m0 = (tile >> 4) << 5;            // 32-row tile
    const int n0 = (tile & 15) << 6;            // 64-col tile
    const int p  = blockIdx.y;

    const __bf16* W    = (p == 0) ? wq : ((p == 1) ? wk : wv);
    const float*  bias = (p == 0) ? bq : ((p == 1) ? bk : bv);

    v8f acc[8];
#pragma unroll
    for (int i = 0; i < 8; ++i) acc[i] = (v8f)0.0f;

    const __bf16* arow0 = xb + (size_t)(m0 + ln) * CC + kb8;
    const __bf16* arow1 = arow0 + (size_t)16 * CC;
    const __bf16* wrow  = W + (size_t)(n0 + ln) * CC + kb8;

    // double-buffered fragment pipeline: loads for k+32 issued before WMMAs
    // that consume k, so s_wait_loadcnt can stay above zero.
    v16bf f0[6], f1[6];
    load_frags6(f0, arow0, arow1, wrow, 0);
    for (int k = 0; k < CC; k += 64) {
        load_frags6(f1, arow0, arow1, wrow, k + 32);
        wmma_tile8(f0, acc);
        if (k + 64 < CC) load_frags6(f0, arow0, arow1, wrow, k + 64);
        wmma_tile8(f1, acc);
    }

#pragma unroll
    for (int mi = 0; mi < 2; ++mi) {
#pragma unroll
        for (int nt = 0; nt < 4; ++nt) {
            const int n = n0 + nt * 16 + ln;
            const int h = n >> 6, d = n & 63;
            const float bvv = bias[n];
#pragma unroll
            for (int i = 0; i < 8; ++i) {
                const int m = m0 + mi * 16 + half + i;
                const int bidx = m >> 11, t = m & (TT - 1);
                const __bf16 val = cvt_bf16(acc[mi * 4 + nt][i] + bvv);
                const size_t bh = (size_t)(bidx * HH + h);
                if (p == 2)       vto[(bh * HD + d) * TT + t] = val;   // V^T
                else if (p == 1)  ko [(bh * TT + t) * HD + d] = val;
                else              qo [(bh * TT + t) * HD + d] = val;
            }
        }
    }
}

// ---------------------------------------------------------------- attention
// Flash-style causal attention. One wave owns 16 query rows.
__global__ __launch_bounds__(256) void csa_attn(
    const __bf16* __restrict__ qb, const __bf16* __restrict__ kbuf,
    const __bf16* __restrict__ vtb, __bf16* __restrict__ yb)
{
    __shared__ __bf16 ldsP[8][16][64];   // per-wave private P staging tile

    const int lane = threadIdx.x & 31;
    const int wid  = threadIdx.x >> 5;
    const int ln   = lane & 15;
    const int kb8  = (lane >> 4) * 8;
    const int half = (lane >> 4) << 3;

    const int h = blockIdx.y, b = blockIdx.z;
    const int q0 = blockIdx.x * 128 + wid * 16;
    const size_t bh = (size_t)(b * HH + h);
    const __bf16* Q  = qb   + bh * (size_t)TT * HD;
    const __bf16* K  = kbuf + bh * (size_t)TT * HD;
    const __bf16* Vt = vtb  + bh * (size_t)HD * TT;

    // Q fragments (16 rows x 64 d) hoisted out of the s-loop
    const __bf16* qrow = Q + (size_t)(q0 + ln) * HD + kb8;
    v16bf aQ0 = frag16(qrow);
    v16bf aQ1 = frag16(qrow + 32);

    v8f accO[4];
#pragma unroll
    for (int dt = 0; dt < 4; ++dt) accO[dt] = (v8f)0.0f;
    float ms[8], ls[8], alphav[8];
#pragma unroll
    for (int i = 0; i < 8; ++i) { ms[i] = -1e30f; ls[i] = 0.0f; }

    const int qEnd = q0 + 15;
    for (int s0 = 0; s0 <= qEnd; s0 += 64) {
        // ---- load all K fragments for this 64-wide s tile, then 8 WMMAs
        v16bf kf0[4], kf1[4];
#pragma unroll
        for (int nt = 0; nt < 4; ++nt) {
            const __bf16* krow = K + (size_t)(s0 + nt * 16 + ln) * HD + kb8;
            kf0[nt] = frag16(krow);
            kf1[nt] = frag16(krow + 32);
        }
        v8f S[4];
#pragma unroll
        for (int nt = 0; nt < 4; ++nt) S[nt] = (v8f)0.0f;
#pragma unroll
        for (int nt = 0; nt < 4; ++nt) {
            S[nt] = wmma_bf16(aQ0, kf0[nt], S[nt]);
            S[nt] = wmma_bf16(aQ1, kf1[nt], S[nt]);
        }

        // ---- issue V fragment loads early; latency hides behind softmax VALU
        v16bf vf0[4], vf1[4];
#pragma unroll
        for (int dt = 0; dt < 4; ++dt) {
            const __bf16* vrow = Vt + (size_t)(dt * 16 + ln) * TT + s0 + kb8;
            vf0[dt] = frag16(vrow);
            vf1[dt] = frag16(vrow + 32);
        }

        // ---- online softmax update (row stats across 16 lanes via shfl_xor)
#pragma unroll
        for (int i = 0; i < 8; ++i) {
            const int row = q0 + half + i;
            float mx = -1e30f;
#pragma unroll
            for (int nt = 0; nt < 4; ++nt) {
                const int col = s0 + nt * 16 + ln;
                float v = S[nt][i] * 0.125f;            // 1/sqrt(64)
                v = (col <= row) ? v : -1e30f;          // causal mask
                S[nt][i] = v;
                mx = fmaxf(mx, v);
            }
#pragma unroll
            for (int msk = 1; msk < 16; msk <<= 1)
                mx = fmaxf(mx, __shfl_xor(mx, msk, 32));
            const float nm = fmaxf(ms[i], mx);
            alphav[i] = __expf(ms[i] - nm);
            ms[i] = nm;
            float rs = 0.0f;
#pragma unroll
            for (int nt = 0; nt < 4; ++nt) {
                const float pp = __expf(S[nt][i] - nm);
                rs += pp;
                ldsP[wid][half + i][nt * 16 + ln] = cvt_bf16(pp);
            }
#pragma unroll
            for (int msk = 1; msk < 16; msk <<= 1)
                rs += __shfl_xor(rs, msk, 32);
            ls[i] = ls[i] * alphav[i] + rs;
        }

        // LDS C-layout -> A-fragment re-layout (intra-wave, in-order DS ops)
        asm volatile("s_wait_dscnt 0" ::: "memory");
        v16bf aP0 = frag16(&ldsP[wid][ln][kb8]);
        v16bf aP1 = frag16(&ldsP[wid][ln][32 + kb8]);

        // ---- O = O*alpha + P V
#pragma unroll
        for (int dt = 0; dt < 4; ++dt) {
#pragma unroll
            for (int i = 0; i < 8; ++i) accO[dt][i] *= alphav[i];
            accO[dt] = wmma_bf16(aP0, vf0[dt], accO[dt]);
            accO[dt] = wmma_bf16(aP1, vf1[dt], accO[dt]);
        }
        asm volatile("" ::: "memory");   // keep next-iter LDS stores behind loads
    }

    // ---- normalize and write y[b,t,h*64+d] (bf16)
#pragma unroll
    for (int dt = 0; dt < 4; ++dt) {
        const int d = dt * 16 + ln;
#pragma unroll
        for (int i = 0; i < 8; ++i) {
            const int t = q0 + half + i;
            const float val = accO[dt][i] / ls[i];
            yb[((size_t)b * TT + t) * CC + h * HD + d] = cvt_bf16(val);
        }
    }
}

// ---------------------------------------------------------------- out proj
__global__ __launch_bounds__(256) void csa_out_proj(
    const __bf16* __restrict__ ybb, const __bf16* __restrict__ wp,
    const float* __restrict__ bp, float* __restrict__ out)
{
    const int lane = threadIdx.x & 31;
    const int wid  = threadIdx.x >> 5;
    const int ln   = lane & 15;
    const int kb8  = (lane >> 4) * 8;
    const int half = (lane >> 4) << 3;

    const int tile = blockIdx.x * 8 + wid;
    const int m0 = (tile >> 4) << 5;
    const int n0 = (tile & 15) << 6;

    v8f acc[8];
#pragma unroll
    for (int i = 0; i < 8; ++i) acc[i] = (v8f)0.0f;

    const __bf16* arow0 = ybb + (size_t)(m0 + ln) * CC + kb8;
    const __bf16* arow1 = arow0 + (size_t)16 * CC;
    const __bf16* wrow  = wp + (size_t)(n0 + ln) * CC + kb8;

    v16bf f0[6], f1[6];
    load_frags6(f0, arow0, arow1, wrow, 0);
    for (int k = 0; k < CC; k += 64) {
        load_frags6(f1, arow0, arow1, wrow, k + 32);
        wmma_tile8(f0, acc);
        if (k + 64 < CC) load_frags6(f0, arow0, arow1, wrow, k + 64);
        wmma_tile8(f1, acc);
    }

#pragma unroll
    for (int mi = 0; mi < 2; ++mi) {
#pragma unroll
        for (int nt = 0; nt < 4; ++nt) {
            const int n = n0 + nt * 16 + ln;
            const float bvv = bp[n];
#pragma unroll
            for (int i = 0; i < 8; ++i) {
                const int m = m0 + mi * 16 + half + i;
                out[(size_t)m * CC + n] = acc[mi * 4 + nt][i] + bvv;
            }
        }
    }
}

// ---------------------------------------------------------------- launch
extern "C" void kernel_launch(void* const* d_in, const int* in_sizes, int n_in,
                              void* d_out, int out_size, void* d_ws, size_t ws_size,
                              hipStream_t stream) {
    (void)in_sizes; (void)n_in; (void)out_size; (void)ws_size;
    const float* x  = (const float*)d_in[0];
    const float* Wk = (const float*)d_in[1];
    const float* bk = (const float*)d_in[2];
    const float* Wq = (const float*)d_in[3];
    const float* bq = (const float*)d_in[4];
    const float* Wv = (const float*)d_in[5];
    const float* bv = (const float*)d_in[6];
    const float* Wp = (const float*)d_in[7];
    const float* bp = (const float*)d_in[8];

    __bf16* ws = (__bf16*)d_ws;
    const size_t NX = (size_t)BB * TT * CC;      // 4,194,304
    const size_t NW = (size_t)CC * CC;           // 1,048,576
    __bf16* xb  = ws;
    __bf16* wqb = xb  + NX;
    __bf16* wkb = wqb + NW;
    __bf16* wvb = wkb + NW;
    __bf16* wpb = wvb + NW;
    __bf16* qb  = wpb + NW;
    __bf16* kb  = qb  + NX;   // [B,H,T,HD] == B*T*C elems
    __bf16* vtb = kb  + NX;   // [B,H,HD,T]
    __bf16* yb  = vtb + NX;   // [B,T,C]

    csa_cvt_f32_bf16<<<(int)((NX + 255) / 256), 256, 0, stream>>>(x,  xb,  (int)NX);
    csa_cvt_f32_bf16<<<(int)((NW + 255) / 256), 256, 0, stream>>>(Wq, wqb, (int)NW);
    csa_cvt_f32_bf16<<<(int)((NW + 255) / 256), 256, 0, stream>>>(Wk, wkb, (int)NW);
    csa_cvt_f32_bf16<<<(int)((NW + 255) / 256), 256, 0, stream>>>(Wv, wvb, (int)NW);
    csa_cvt_f32_bf16<<<(int)((NW + 255) / 256), 256, 0, stream>>>(Wp, wpb, (int)NW);

    csa_qkv_proj<<<dim3(256, 3, 1), 256, 0, stream>>>(
        xb, wqb, wkb, wvb, bq, bk, bv, qb, kb, vtb);

    csa_attn<<<dim3(TT / 128, HH, BB), 256, 0, stream>>>(qb, kb, vtb, yb);

    csa_out_proj<<<256, 256, 0, stream>>>(yb, wpb, bp, (float*)d_out);
}
